// MSACol_20194936225881
// MI455X (gfx1250) — matrix operand order repaired and checked by reading, not compiled, not run on hardware
//
#include <hip/hip_runtime.h>
#include <math.h>

// ---------------------------------------------------------------------------
// MSA column (global) attention for MI455X / gfx1250, wave32 + WMMA f16.
// Pipeline: [0] weight transpose->f16  [1] LN + QKVG projection (WMMA)
//           [2] logits per head (WMMA, K=12288)  [3] softmax  [4] P@V * gate
//           [5] output projection (WMMA) + bias.
// ---------------------------------------------------------------------------

typedef _Float16 half_t;
typedef half_t v8h  __attribute__((ext_vector_type(8)));
typedef half_t v16h __attribute__((ext_vector_type(16)));
typedef float  v8f  __attribute__((ext_vector_type(8)));

#define N_SEQ 512
#define L_RES 384
#define D_MODEL 256
#define N_HEAD 8
#define C_HEAD 32
#define NL (N_SEQ * L_RES)          // 196608
#define KHC (L_RES * C_HEAD)        // 12288, logits contraction length

// workspace byte offsets (total ~493 MiB)
#define Q_OFF   ((size_t)0)                  // [8][512][12288] f16
#define K_OFF   ((size_t)100663296)
#define G_OFF   ((size_t)201326592)
#define VT_OFF  ((size_t)301989888)          // [8][12288][512] f16 (transposed)
#define OG_OFF  ((size_t)402653184)          // [196608][256] f16 (o*g row-major)
#define ATT_OFF ((size_t)503316480)          // [8][512][512] f32
#define P_OFF   ((size_t)511705088)          // [8][512][512] f16
#define WT_OFF  ((size_t)515899392)          // [1024][256] f16 (Wq|Wk|Wv|Wg)^T
#define WOT_OFF ((size_t)516423680)          // [256][256] f16 Wo^T

// A operand, f16 16x16x32: lane holds row (lane&15); 16 consecutive K values
// starting at k0 + 16*(lane>>4). Both halves are 16B-aligned contiguous loads.
__device__ __forceinline__ v16h load_a16(const half_t* p) {
  v8h lo = *reinterpret_cast<const v8h*>(p);
  v8h hi = *reinterpret_cast<const v8h*>(p + 8);
  return __builtin_shufflevector(lo, hi, 0,1,2,3,4,5,6,7,8,9,10,11,12,13,14,15);
}
// B operand from a BT (N-major) matrix: lane holds col (lane&15), 16
// consecutive K values starting at k0 + 16*(lane>>4) -> 32B contiguous.
__device__ __forceinline__ v16h load_b16(const half_t* p) {
  v8h lo = *reinterpret_cast<const v8h*>(p);
  v8h hi = *reinterpret_cast<const v8h*>(p + 8);
  return __builtin_shufflevector(lo, hi, 0,1,2,3,4,5,6,7,8,9,10,11,12,13,14,15);
}
__device__ __forceinline__ v8f wmma16(v16h a, v16h b, v8f c) {
  return __builtin_amdgcn_wmma_f32_16x16x32_f16(false, a, false, b, (short)0, c,
                                                false, false);
}

// ---------------------------------------------------------------------------
// Kernel 0: transpose + convert weights to f16.  wt = [1024 cols][256 k],
// col = p*256 + n over {Wq,Wk,Wv,Wg}; wot = [256 cols][256 k] from Wo.
// ---------------------------------------------------------------------------
__global__ __launch_bounds__(256) void k0_wtrans(
    const float* __restrict__ Wq, const float* __restrict__ Wk,
    const float* __restrict__ Wv, const float* __restrict__ Wg,
    const float* __restrict__ Wo, half_t* __restrict__ wt,
    half_t* __restrict__ wot) {
  int e = blockIdx.x * 256 + threadIdx.x;
  if (e < 262144) {
    int col = e >> 8, kk = e & 255;
    int p = col >> 8, n = col & 255;
    const float* W = (p == 0) ? Wq : (p == 1) ? Wk : (p == 2) ? Wv : Wg;
    wt[e] = (half_t)W[kk * 256 + n];
  } else {
    int e2 = e - 262144;
    int n = e2 >> 8, kk = e2 & 255;
    wot[e2] = (half_t)Wo[kk * 256 + n];
  }
}

// ---------------------------------------------------------------------------
// Kernel 1: fused LayerNorm + 4 projections. Block = 256 thr (8 waves),
// one M-tile of 16 rows of m[NL][256]. Each wave owns 8 N-tiles (128 cols
// of the 1024 concatenated outputs), K-loop of 8 WMMAs per tile.
// ---------------------------------------------------------------------------
__global__ __launch_bounds__(256) void k1_ln_proj(
    const float* __restrict__ m, const float* __restrict__ gamma,
    const float* __restrict__ beta, const float* __restrict__ bg,
    const half_t* __restrict__ wt, half_t* __restrict__ q_out,
    half_t* __restrict__ k_out, half_t* __restrict__ g_out,
    half_t* __restrict__ vT_out) {
  __shared__ float  sm[16 * 256];
  __shared__ half_t smn[16 * 256];

  const int tid = threadIdx.x;
  const int wave = tid >> 5, lane = tid & 31;
  const int lrow = lane & 15, lhi = lane >> 4;
  const int r0 = blockIdx.x * 16;

  // stage m rows into LDS (float4 vectorized)
  const float4* gsrc = reinterpret_cast<const float4*>(m + (size_t)r0 * 256);
  float4* lds4 = reinterpret_cast<float4*>(sm);
#pragma unroll
  for (int j = 0; j < 4; ++j) lds4[tid + j * 256] = gsrc[tid + j * 256];
  __syncthreads();

  // layernorm: wave w handles rows 2w, 2w+1; wave-shuffle reduction
#pragma unroll
  for (int rr = 0; rr < 2; ++rr) {
    int row = wave * 2 + rr;
    float s = 0.f, sq = 0.f;
#pragma unroll
    for (int j = 0; j < 8; ++j) {
      float x = sm[row * 256 + lane + 32 * j];
      s += x; sq += x * x;
    }
#pragma unroll
    for (int off = 16; off > 0; off >>= 1) {
      s  += __shfl_xor(s, off, 32);
      sq += __shfl_xor(sq, off, 32);
    }
    float mu = s * (1.0f / 256.0f);
    float var = sq * (1.0f / 256.0f) - mu * mu;
    float rstd = rsqrtf(var + 1e-5f);
#pragma unroll
    for (int j = 0; j < 8; ++j) {
      int col = lane + 32 * j;
      float x = sm[row * 256 + col];
      smn[row * 256 + col] = (half_t)((x - mu) * rstd * gamma[col] + beta[col]);
    }
  }
  __syncthreads();

  // GEMM: A = smn (LDS), B = wt (global BT rows, contiguous)
  v8f acc[8] = {};
#pragma unroll
  for (int kk = 0; kk < 8; ++kk) {
    const int k0 = kk * 32;
    v16h a = load_a16(&smn[lrow * 256 + k0 + 16 * lhi]);
#pragma unroll
    for (int t = 0; t < 8; ++t) {
      int col = (wave * 8 + t) * 16 + lrow;
      v16h b = load_b16(wt + (size_t)col * 256 + k0 + 16 * lhi);
      acc[t] = wmma16(a, b, acc[t]);
    }
  }

  // epilogue: scatter to per-head layouts (p uniform per wave: p = wave/2)
  const int p = wave >> 1;
#pragma unroll
  for (int t = 0; t < 8; ++t) {
    int col = (wave * 8 + t) * 16 + lrow;
    int colp = col & 255;
    int h = colp >> 5, c = colp & 31;
#pragma unroll
    for (int i = 0; i < 8; ++i) {
      int r = r0 + i + 8 * lhi;
      int s = r / 384, l = r - s * 384;
      float val = acc[t][i];
      size_t hs_idx = (size_t)(h * 512 + s) * KHC + l * 32 + c;
      if (p == 0) {
        q_out[hs_idx] = (half_t)val;
      } else if (p == 1) {
        k_out[hs_idx] = (half_t)val;
      } else if (p == 2) {
        vT_out[(size_t)(h * KHC + l * 32 + c) * 512 + s] = (half_t)val;
      } else {
        float gv = 1.0f / (1.0f + __expf(-(val + bg[colp])));
        g_out[hs_idx] = (half_t)gv;
      }
    }
  }
}

// ---------------------------------------------------------------------------
// Kernel 2: att[h][s][t] = scale * q_h @ k_h^T, K = 12288. Block = 128 thr
// (4 waves), block covers 64x64, each wave a 2x2 register block of tiles.
// ---------------------------------------------------------------------------
__global__ __launch_bounds__(128) void k2_logits(
    const half_t* __restrict__ q_ws, const half_t* __restrict__ k_ws,
    float* __restrict__ att_ws) {
  const int bid = blockIdx.x;
  const int h = bid >> 6, rem = bid & 63;
  const int sblk = rem >> 3, tblk = rem & 7;
  const int wave = threadIdx.x >> 5, lane = threadIdx.x & 31;
  const int lrow = lane & 15, lhi = lane >> 4;
  const int s_base = sblk * 64 + (wave >> 1) * 32;
  const int t_base = tblk * 64 + (wave & 1) * 32;
  const half_t* qh = q_ws + (size_t)h * 512 * KHC;
  const half_t* kh = k_ws + (size_t)h * 512 * KHC;

  v8f acc[2][2] = {};
  for (int k0 = 0; k0 < KHC; k0 += 32) {
    v16h a[2], b[2];
#pragma unroll
    for (int i2 = 0; i2 < 2; ++i2)
      a[i2] = load_a16(qh + (size_t)(s_base + i2 * 16 + lrow) * KHC + k0 + 16 * lhi);
#pragma unroll
    for (int j2 = 0; j2 < 2; ++j2)
      b[j2] = load_b16(kh + (size_t)(t_base + j2 * 16 + lrow) * KHC + k0 + 16 * lhi);
#pragma unroll
    for (int i2 = 0; i2 < 2; ++i2)
#pragma unroll
      for (int j2 = 0; j2 < 2; ++j2)
        acc[i2][j2] = wmma16(a[i2], b[j2], acc[i2][j2]);
  }

  const float scale = 0.009021097868f;  // (1/sqrt(32))/sqrt(384)
  float* att = att_ws + (size_t)h * 512 * 512;
#pragma unroll
  for (int i2 = 0; i2 < 2; ++i2)
#pragma unroll
    for (int j2 = 0; j2 < 2; ++j2)
#pragma unroll
      for (int i = 0; i < 8; ++i) {
        int s = s_base + i2 * 16 + i + 8 * lhi;
        int t = t_base + j2 * 16 + lrow;
        att[(size_t)s * 512 + t] = acc[i2][j2][i] * scale;
      }
}

// ---------------------------------------------------------------------------
// Kernel 3: softmax over t for each (h,s) row of 512; one wave per row.
// ---------------------------------------------------------------------------
__global__ __launch_bounds__(256) void k3_softmax(
    const float* __restrict__ att_ws, half_t* __restrict__ p_ws) {
  const int wave = threadIdx.x >> 5, lane = threadIdx.x & 31;
  const int row = blockIdx.x * 8 + wave;
  const float* a = att_ws + (size_t)row * 512;
  float v[16];
  float mx = -3.4e38f;
#pragma unroll
  for (int j = 0; j < 16; ++j) {
    v[j] = a[lane + 32 * j];
    mx = fmaxf(mx, v[j]);
  }
#pragma unroll
  for (int off = 16; off > 0; off >>= 1) mx = fmaxf(mx, __shfl_xor(mx, off, 32));
  float ssum = 0.f;
#pragma unroll
  for (int j = 0; j < 16; ++j) {
    v[j] = __expf(v[j] - mx);
    ssum += v[j];
  }
#pragma unroll
  for (int off = 16; off > 0; off >>= 1) ssum += __shfl_xor(ssum, off, 32);
  float inv = 1.0f / ssum;
  half_t* pr = p_ws + (size_t)row * 512;
#pragma unroll
  for (int j = 0; j < 16; ++j) pr[lane + 32 * j] = (half_t)(v[j] * inv);
}

// ---------------------------------------------------------------------------
// Kernel 4: o_h = P_h @ v_h (K = 512 over t), gate with g, scatter to og
// row-major [s*384+l][h*32+c]. Block = 128 thr covers 64x64 (s x lc).
// ---------------------------------------------------------------------------
__global__ __launch_bounds__(128) void k4_attnout(
    const half_t* __restrict__ p_ws, const half_t* __restrict__ vT_ws,
    const half_t* __restrict__ g_ws, half_t* __restrict__ og_ws) {
  const int bid = blockIdx.x;
  const int h = bid / 1536, rem = bid % 1536;
  const int sblk = rem / 192, cblk = rem % 192;
  const int wave = threadIdx.x >> 5, lane = threadIdx.x & 31;
  const int lrow = lane & 15, lhi = lane >> 4;
  const int s_base = sblk * 64 + (wave >> 1) * 32;
  const int c_base = cblk * 64 + (wave & 1) * 32;  // lc coordinate
  const half_t* ph = p_ws + (size_t)h * 512 * 512;
  const half_t* vh = vT_ws + (size_t)h * KHC * 512;

  v8f acc[2][2] = {};
  for (int k0 = 0; k0 < 512; k0 += 32) {
    v16h a[2], b[2];
#pragma unroll
    for (int i2 = 0; i2 < 2; ++i2)
      a[i2] = load_a16(ph + (size_t)(s_base + i2 * 16 + lrow) * 512 + k0 + 16 * lhi);
#pragma unroll
    for (int j2 = 0; j2 < 2; ++j2)
      b[j2] = load_b16(vh + (size_t)(c_base + j2 * 16 + lrow) * 512 + k0 + 16 * lhi);
#pragma unroll
    for (int i2 = 0; i2 < 2; ++i2)
#pragma unroll
      for (int j2 = 0; j2 < 2; ++j2)
        acc[i2][j2] = wmma16(a[i2], b[j2], acc[i2][j2]);
  }

  const half_t* gh = g_ws + (size_t)h * 512 * KHC;
#pragma unroll
  for (int i2 = 0; i2 < 2; ++i2)
#pragma unroll
    for (int j2 = 0; j2 < 2; ++j2)
#pragma unroll
      for (int i = 0; i < 8; ++i) {
        int s = s_base + i2 * 16 + i + 8 * lhi;
        int lc = c_base + j2 * 16 + lrow;
        float gv = (float)gh[(size_t)s * KHC + lc];
        float o = acc[i2][j2][i] * gv;
        int l = lc >> 5, c = lc & 31;
        og_ws[(size_t)(s * 384 + l) * 256 + h * 32 + c] = (half_t)o;
      }
}

// ---------------------------------------------------------------------------
// Kernel 5: out = og @ Wo^T-layout + bo.  M = 196608, N = 256, K = 256.
// ---------------------------------------------------------------------------
__global__ __launch_bounds__(128) void k5_outproj(
    const half_t* __restrict__ og_ws, const half_t* __restrict__ wot,
    const float* __restrict__ bo, float* __restrict__ out) {
  const int bid = blockIdx.x;
  const int mblk = bid >> 2, nblk = bid & 3;
  const int wave = threadIdx.x >> 5, lane = threadIdx.x & 31;
  const int lrow = lane & 15, lhi = lane >> 4;
  const int r_base = mblk * 64 + (wave >> 1) * 32;
  const int c_base = nblk * 64 + (wave & 1) * 32;

  v8f acc[2][2] = {};
#pragma unroll
  for (int kk = 0; kk < 8; ++kk) {
    const int k0 = kk * 32;
    v16h a[2], b[2];
#pragma unroll
    for (int i2 = 0; i2 < 2; ++i2)
      a[i2] = load_a16(og_ws + (size_t)(r_base + i2 * 16 + lrow) * 256 + k0 + 16 * lhi);
#pragma unroll
    for (int j2 = 0; j2 < 2; ++j2)
      b[j2] = load_b16(wot + (size_t)(c_base + j2 * 16 + lrow) * 256 + k0 + 16 * lhi);
#pragma unroll
    for (int i2 = 0; i2 < 2; ++i2)
#pragma unroll
      for (int j2 = 0; j2 < 2; ++j2)
        acc[i2][j2] = wmma16(a[i2], b[j2], acc[i2][j2]);
  }

#pragma unroll
  for (int i2 = 0; i2 < 2; ++i2)
#pragma unroll
    for (int j2 = 0; j2 < 2; ++j2)
#pragma unroll
      for (int i = 0; i < 8; ++i) {
        int r = r_base + i2 * 16 + i + 8 * lhi;
        int col = c_base + j2 * 16 + lrow;
        out[(size_t)r * 256 + col] = acc[i2][j2][i] + bo[col];
      }
}

// ---------------------------------------------------------------------------
extern "C" void kernel_launch(void* const* d_in, const int* in_sizes, int n_in,
                              void* d_out, int out_size, void* d_ws,
                              size_t ws_size, hipStream_t stream) {
  const float* m     = (const float*)d_in[0];
  const float* gamma = (const float*)d_in[1];
  const float* beta  = (const float*)d_in[2];
  const float* Wq    = (const float*)d_in[3];
  const float* Wk    = (const float*)d_in[4];
  const float* Wv    = (const float*)d_in[5];
  const float* Wg    = (const float*)d_in[6];
  const float* bg    = (const float*)d_in[7];
  const float* Wo    = (const float*)d_in[8];
  const float* bo    = (const float*)d_in[9];
  float* out = (float*)d_out;

  char* ws = (char*)d_ws;
  half_t* q_ws  = (half_t*)(ws + Q_OFF);
  half_t* k_ws  = (half_t*)(ws + K_OFF);
  half_t* g_ws  = (half_t*)(ws + G_OFF);
  half_t* vT_ws = (half_t*)(ws + VT_OFF);
  half_t* og_ws = (half_t*)(ws + OG_OFF);
  float*  att_ws = (float*)(ws + ATT_OFF);
  half_t* p_ws  = (half_t*)(ws + P_OFF);
  half_t* wt    = (half_t*)(ws + WT_OFF);
  half_t* wot   = (half_t*)(ws + WOT_OFF);

  k0_wtrans<<<1280, 256, 0, stream>>>(Wq, Wk, Wv, Wg, Wo, wt, wot);
  k1_ln_proj<<<NL / 16, 256, 0, stream>>>(m, gamma, beta, bg, wt,
                                          q_ws, k_ws, g_ws, vT_ws);
  k2_logits<<<512, 128, 0, stream>>>(q_ws, k_ws, att_ws);
  k3_softmax<<<512, 256, 0, stream>>>(att_ws, p_ws);
  k4_attnout<<<12288, 128, 0, stream>>>(p_ws, vT_ws, g_ws, og_ws);
  k5_outproj<<<12288, 128, 0, stream>>>(og_ws, wot, bo, out);
}
